// ContinuousRecurrentCore_17944373362960
// MI455X (gfx1250) — matrix operand order repaired and checked by reading, not compiled, not run on hardware
//
#include <hip/hip_runtime.h>
#include <hip/hip_bf16.h>
#include <stdint.h>

typedef __attribute__((ext_vector_type(16))) __bf16 v16bf;
typedef __attribute__((ext_vector_type(8)))  float  v8f;

#define NL      4
#define D       256
#define K2      512           // 2*D (concat K) == N (tau|forcing)
#define TOKENS  8192          // B*S = 8*1024
#define TILE_M  64
#define NBLK    (TOKENS / TILE_M)
#define TMAXI   50
#define DTC     0.05f
#define EPSC    1e-3f

// ---------------------------------------------------------------- utilities
__global__ __launch_bounds__(256) void zero_fill_kernel(uint4* p, int n) {
  int i = blockIdx.x * 256 + threadIdx.x;
  if (i < n) p[i] = uint4{0u, 0u, 0u, 0u};
}

__global__ void init_ctrl_kernel(unsigned int* ctrl) {
  ctrl[0] = 0u;                 // delta (float bits)
  ctrl[1] = 0u;                 // done
  ((int*)ctrl)[2] = TMAXI;      // result
}

// Pack combined weight matrix [NL][N=512][K=512] into WMMA B-fragment order.
// Fragment (layer, nsub, kst) = 32x16 bf16, per lane 16 contiguous bf16:
//   n = nsub*16 + (lane&15),  k = kst*32 + i + 16*(lane>=16)   (contiguous k order, as in A frags)
// N<256 -> tau rows (Wt[d][k], k over [cur|h]); N>=256 -> forcing rows
//   (k<256: Wi[e][k], k>=256: Ws[e][k-256]).
__global__ __launch_bounds__(256) void pack_weights_kernel(
    const float* __restrict__ Ws, const float* __restrict__ Wi,
    const float* __restrict__ Wt, __bf16* __restrict__ wpack) {
  int idx   = blockIdx.x * 256 + threadIdx.x;   // 2^20 entries
  int i     =  idx        & 15;
  int lane  = (idx >> 4)  & 31;
  int kst   = (idx >> 9)  & 15;
  int nsub  = (idx >> 13) & 31;
  int layer = (idx >> 18) & 3;
  int n = nsub * 16 + (lane & 15);
  int k = kst * 32 + i + ((lane >> 4) << 4);
  float v;
  if (n < D) {
    v = Wt[((size_t)layer * D + n) * K2 + k];
  } else {
    int e = n - D;
    if (k < D) v = Wi[((size_t)layer * D + e) * D + k];
    else       v = Ws[((size_t)layer * D + e) * D + (k - D)];
  }
  wpack[idx] = (__bf16)v;
}

__global__ void pack_bias_kernel(const float* bs, const float* bi,
                                 const float* bt, float* biasc) {
  int idx = blockIdx.x * 256 + threadIdx.x;     // NL*512
  if (idx >= NL * K2) return;
  int layer = idx >> 9, n = idx & 511;
  biasc[idx] = (n < D) ? bt[layer * D + n]
                       : (bs[layer * D + (n - D)] + bi[layer * D + (n - D)]);
}

__global__ __launch_bounds__(256) void f32_to_bf16_kernel(
    const float* __restrict__ in, __bf16* __restrict__ out, int n) {
  int i = blockIdx.x * 256 + threadIdx.x;
  if (i < n) out[i] = (__bf16)in[i];
}

__global__ void begin_iter_kernel(unsigned int* ctrl) { ctrl[0] = 0u; }

__global__ void end_iter_kernel(unsigned int* ctrl, int t) {
  if (ctrl[1]) return;
  if (__uint_as_float(ctrl[0]) < EPSC) { ctrl[1] = 1u; ((int*)ctrl)[2] = t; }
}

// ------------------------------------------------- fused GEMM + dynamics
// One workgroup: 64 tokens x full N=512, K=512 in 16 steps of 32.
// Wave w owns n-subtiles {2w, 2w+1, 2w+16, 2w+17} (matched tau/forcing pairs)
// and all 4 m-subtiles -> 16 f32 accumulators (128 VGPRs).
__global__ __launch_bounds__(256) void layer_kernel(
    const __bf16* __restrict__ wpack,   // packed weights, all layers
    const float*  __restrict__ biasc,   // [NL][512]
    const __bf16* __restrict__ curb,    // cur_in bf16 [TOKENS][D]
    __bf16*       __restrict__ hb,      // state bf16  [TOKENS][D] (in/out)
    float*        __restrict__ hf,      // state f32   [TOKENS][D] (in/out)
    const unsigned int* __restrict__ donep,
    unsigned int* __restrict__ deltap,
    int layer) {
  if (*donep) return;                       // uniform: frozen once converged

  __shared__ __align__(16) __bf16 tileA[TILE_M * 32];
  __shared__ float red[256];

  const int tid     = threadIdx.x;
  const int wave    = tid >> 5;
  const int lane    = tid & 31;
  const int laneHi  = lane >> 4;
  const int lane15  = lane & 15;
  const int tokBase = blockIdx.x * TILE_M;

  v8f zero = {0.f, 0.f, 0.f, 0.f, 0.f, 0.f, 0.f, 0.f};
  v8f acc[4][4];
#pragma unroll
  for (int a = 0; a < 4; ++a)
#pragma unroll
    for (int b = 0; b < 4; ++b) acc[a][b] = zero;

  const __bf16* wl = wpack + (size_t)layer * (32 * 16 * 512);

  for (int kst = 0; kst < 16; ++kst) {
    const int kg = kst * 32;
    const __bf16* src = (kg < D) ? (curb + kg) : (hb + (kg - D));
    {   // cooperative stage: 64 rows x 32 k of bf16 (16B per thread)
      int row = tid >> 2, ch = tid & 3;
      uint4 v = *(const uint4*)(src + (size_t)(tokBase + row) * D + ch * 8);
      *(uint4*)(tileA + row * 32 + ch * 8) = v;
    }
    __syncthreads();

    v16bf afr[4];                       // A frags: K = 0..15 + 16*hi
#pragma unroll
    for (int im = 0; im < 4; ++im) {
      const __bf16* p = tileA + (im * 16 + lane15) * 32 + laneHi * 16;
      ((uint4*)&afr[im])[0] = *(const uint4*)p;
      ((uint4*)&afr[im])[1] = *(const uint4*)(p + 8);
    }

#pragma unroll
    for (int jn = 0; jn < 4; ++jn) {
      const int nsub = 2 * wave + (jn & 1) + ((jn >> 1) << 4);
      const __bf16* bp = wl + ((size_t)(nsub * 16 + kst) << 9) + lane * 16;
      __builtin_prefetch(bp + 512, 0, 0);       // next k-step of this strip
      v16bf bfr;
      ((uint4*)&bfr)[0] = *(const uint4*)bp;
      ((uint4*)&bfr)[1] = *(const uint4*)(bp + 8);
#pragma unroll
      for (int im = 0; im < 4; ++im)
        acc[im][jn] = __builtin_amdgcn_wmma_f32_16x16x32_bf16(
            false, afr[im], false, bfr, (short)0, acc[im][jn], false, false);
    }
    __syncthreads();                    // tileA consumed before next stage
  }

  // ------------- fused pointwise dynamics (tau = acc[.][0..1], forcing +2)
  float dmax = 0.f;
  const int dcol0 = wave * 32;
#pragma unroll
  for (int im = 0; im < 4; ++im) {
    const int tok0 = tokBase + im * 16 + laneHi * 8;
#pragma unroll
    for (int jp = 0; jp < 2; ++jp) {
      const int d   = dcol0 + jp * 16 + lane15;
      const float btc = biasc[layer * K2 + d];
      const float bfc = biasc[layer * K2 + D + d];
      v8f tA = acc[im][jp];
      v8f fA = acc[im][jp + 2];
#pragma unroll
      for (int r = 0; r < 8; ++r) {
        float tau_pre  = tA[r] + btc;
        float forc_pre = fA[r] + bfc;
        float tau     = 1.f / (1.f + __expf(-tau_pre));
        float forcing = tanhf(forc_pre);
        size_t off = (size_t)(tok0 + r) * D + d;
        float hp = hf[off];
        float hn = hp + DTC * (-hp / (tau + 1e-6f) + forcing);
        hn = fminf(10.f, fmaxf(-10.f, hn));
        hf[off] = hn;
        hb[off] = (__bf16)hn;
        dmax = fmaxf(dmax, fabsf(hn - hp));
      }
    }
  }

  red[tid] = dmax;
  __syncthreads();
  for (int s = 128; s > 0; s >>= 1) {
    if (tid < s) red[tid] = fmaxf(red[tid], red[tid + s]);
    __syncthreads();
  }
  if (tid == 0) atomicMax(deltap, __float_as_uint(red[0]));
}

__global__ __launch_bounds__(256) void finalize_kernel(
    const float* __restrict__ hf3, const unsigned int* __restrict__ ctrl,
    float* __restrict__ out) {
  int i = blockIdx.x * 256 + threadIdx.x;
  const int n = TOKENS * D;
  if (i < n) out[i] = hf3[i];
  if (i == 0) out[n] = (float)((const int*)ctrl)[2];
}

// ---------------------------------------------------------------- launcher
extern "C" void kernel_launch(void* const* d_in, const int* in_sizes, int n_in,
                              void* d_out, int out_size, void* d_ws, size_t ws_size,
                              hipStream_t stream) {
  const float* x  = (const float*)d_in[0];
  const float* Ws = (const float*)d_in[1];
  const float* bs = (const float*)d_in[2];
  const float* Wi = (const float*)d_in[3];
  const float* bi = (const float*)d_in[4];
  const float* Wt = (const float*)d_in[5];
  const float* bt = (const float*)d_in[6];

  char* ws = (char*)d_ws;
  const size_t SZ_WP = (size_t)NL * K2 * K2 * 2;        //  2 MB packed weights
  const size_t SZ_BC = (size_t)NL * K2 * 4;             //  8 KB bias
  const size_t SZ_XB = (size_t)TOKENS * D * 2;          //  4 MB x bf16
  const size_t SZ_SF = (size_t)NL * TOKENS * D * 4;     // 32 MB states f32
  const size_t SZ_SB = (size_t)NL * TOKENS * D * 2;     // 16 MB states bf16
  const size_t OFF_WP = 0;
  const size_t OFF_BC = OFF_WP + SZ_WP;
  const size_t OFF_XB = OFF_BC + SZ_BC;
  const size_t OFF_SF = OFF_XB + SZ_XB;
  const size_t OFF_SB = OFF_SF + SZ_SF;
  const size_t OFF_CT = OFF_SB + SZ_SB;

  __bf16* wpack      = (__bf16*)(ws + OFF_WP);
  float*  biasc      = (float*)(ws + OFF_BC);
  __bf16* xb         = (__bf16*)(ws + OFF_XB);
  float*  sf         = (float*)(ws + OFF_SF);
  __bf16* sb         = (__bf16*)(ws + OFF_SB);
  unsigned int* ctrl = (unsigned int*)(ws + OFF_CT);

  // deterministic re-init every call (states + bf16 mirror are contiguous)
  int n16 = (int)((SZ_SF + SZ_SB) / 16);
  zero_fill_kernel<<<(n16 + 255) / 256, 256, 0, stream>>>((uint4*)(ws + OFF_SF), n16);
  init_ctrl_kernel<<<1, 1, 0, stream>>>(ctrl);
  pack_weights_kernel<<<(NL * 32 * 16 * 512) / 256, 256, 0, stream>>>(Ws, Wi, Wt, wpack);
  pack_bias_kernel<<<8, 256, 0, stream>>>(bs, bi, bt, biasc);
  f32_to_bf16_kernel<<<(TOKENS * D) / 256, 256, 0, stream>>>(x, xb, TOKENS * D);

  for (int t = 0; t < TMAXI; ++t) {
    begin_iter_kernel<<<1, 1, 0, stream>>>(ctrl);
    for (int l = 0; l < NL; ++l) {
      const __bf16* curb = (l == 0) ? xb : (sb + (size_t)(l - 1) * TOKENS * D);
      layer_kernel<<<NBLK, 256, 0, stream>>>(
          wpack, biasc, curb,
          sb + (size_t)l * TOKENS * D,
          sf + (size_t)l * TOKENS * D,
          ctrl + 1, ctrl + 0, l);
    }
    end_iter_kernel<<<1, 1, 0, stream>>>(ctrl, t);
  }

  finalize_kernel<<<(TOKENS * D) / 256, 256, 0, stream>>>(
      sf + (size_t)(NL - 1) * TOKENS * D, ctrl, (float*)d_out);
}
